// LSTMClassifier_55190329753762
// MI455X (gfx1250) — matrix operand order repaired and checked by reading, not compiled, not run on hardware
//
#include <hip/hip_runtime.h>
#include <hip/hip_bf16.h>
#include <math.h>

// ---------------------------------------------------------------------------
// LSTM classifier for MI455X (gfx1250, wave32, WMMA, bf16 A/B + f32 accum).
//
// Path A (ws_size >= 589 MiB):
//   K0: W_ih/W_hh f32->bf16       K1: embedding gather f32->bf16
//   K2: zero h/c                  K3: xg = x @ W_ih^T  (one big parallel GEMM)
//   K4 x 512: step  gates = xg_t + b + h @ W_hh^T, K=1024 split 2-way across
//             wave pairs (LDS reduction) -> 1024 waves, half-length chains
//   K5: classifier
// Path B (small ws): fused step with K=1536 (input projection inside step).
//
// K-loops are manually software-pipelined with a two-buffer ping-pong
// (#pragma unroll 1 so the unroller cannot dissolve the structure): loads for
// chunks c,c+1 are issued before the WMMAs consuming chunks c-1,c.
// ---------------------------------------------------------------------------

typedef __bf16 bf16;
typedef bf16  v16bf __attribute__((ext_vector_type(16)));
typedef float v8f   __attribute__((ext_vector_type(8)));

static constexpr int B    = 128;
static constexpr int S    = 512;
static constexpr int E    = 512;
static constexpr int H    = 1024;
static constexpr int G4H  = 4 * H;   // 4096
static constexpr int OUTN = 8;

// --- WMMA fragment loaders (contiguous k order, k = sub*16 + elem, K contiguous in memory) ----
// A 16x32 bf16: lane l -> row m=l&15; sub=l>>4: elems 0..15 = K[sub*16 .. +16)
static __device__ inline v16bf load_a_frag(const bf16* __restrict__ row, int k0, int sub) {
    union { v16bf v; uint4 q[2]; } u;
    u.q[0] = *reinterpret_cast<const uint4*>(row + k0 + sub * 16);
    u.q[1] = *reinterpret_cast<const uint4*>(row + k0 + sub * 16 + 8);
    return u.v;
}
// B 32x16 bf16: lane l -> col n=l&15; sub=l>>4: elems 0..15 = K[sub*16 .. +16)
static __device__ inline v16bf load_b_frag(const bf16* __restrict__ row, int k0, int sub) {
    union { v16bf v; uint4 q[2]; } u;
    const bf16* p = row + k0 + sub * 16;
    u.q[0] = *reinterpret_cast<const uint4*>(p);
    u.q[1] = *reinterpret_cast<const uint4*>(p + 8);
    return u.v;
}

#define WMMA_BF16(a, b, c) \
    __builtin_amdgcn_wmma_f32_16x16x32_bf16(false, (a), false, (b), (short)0, (c), false, false)

// Two-buffer ping-pong K-loop over 4 B-streams sharing one A-stream.
// Requires K a multiple of 64 (even number of 32-chunks, >= 2).
static __device__ inline void gemm_k_pipe(
    const bf16* __restrict__ arow,
    const bf16* __restrict__ b0r, const bf16* __restrict__ b1r,
    const bf16* __restrict__ b2r, const bf16* __restrict__ b3r,
    int K, int sub, v8f& acc0, v8f& acc1, v8f& acc2, v8f& acc3)
{
    v16bf ax, bx0, bx1, bx2, bx3;   // set X
    v16bf ay, by0, by1, by2, by3;   // set Y
    ax  = load_a_frag(arow, 0, sub);
    bx0 = load_b_frag(b0r, 0, sub);
    bx1 = load_b_frag(b1r, 0, sub);
    bx2 = load_b_frag(b2r, 0, sub);
    bx3 = load_b_frag(b3r, 0, sub);
    #pragma unroll 1
    for (int k0 = 32; k0 < K - 32; k0 += 64) {
        ay  = load_a_frag(arow, k0, sub);
        by0 = load_b_frag(b0r, k0, sub);
        by1 = load_b_frag(b1r, k0, sub);
        by2 = load_b_frag(b2r, k0, sub);
        by3 = load_b_frag(b3r, k0, sub);
        acc0 = WMMA_BF16(ax, bx0, acc0);
        acc1 = WMMA_BF16(ax, bx1, acc1);
        acc2 = WMMA_BF16(ax, bx2, acc2);
        acc3 = WMMA_BF16(ax, bx3, acc3);
        ax  = load_a_frag(arow, k0 + 32, sub);
        bx0 = load_b_frag(b0r, k0 + 32, sub);
        bx1 = load_b_frag(b1r, k0 + 32, sub);
        bx2 = load_b_frag(b2r, k0 + 32, sub);
        bx3 = load_b_frag(b3r, k0 + 32, sub);
        acc0 = WMMA_BF16(ay, by0, acc0);
        acc1 = WMMA_BF16(ay, by1, acc1);
        acc2 = WMMA_BF16(ay, by2, acc2);
        acc3 = WMMA_BF16(ay, by3, acc3);
    }
    // X holds chunk K-64; chunk K-32 still unloaded.
    ay  = load_a_frag(arow, K - 32, sub);
    by0 = load_b_frag(b0r, K - 32, sub);
    by1 = load_b_frag(b1r, K - 32, sub);
    by2 = load_b_frag(b2r, K - 32, sub);
    by3 = load_b_frag(b3r, K - 32, sub);
    acc0 = WMMA_BF16(ax, bx0, acc0);
    acc1 = WMMA_BF16(ax, bx1, acc1);
    acc2 = WMMA_BF16(ax, bx2, acc2);
    acc3 = WMMA_BF16(ax, bx3, acc3);
    acc0 = WMMA_BF16(ay, by0, acc0);
    acc1 = WMMA_BF16(ay, by1, acc1);
    acc2 = WMMA_BF16(ay, by2, acc2);
    acc3 = WMMA_BF16(ay, by3, acc3);
}

static __device__ inline float sigf(float x) { return 1.0f / (1.0f + __expf(-x)); }

// --------------------------- K0: f32 -> bf16 -------------------------------
__global__ void cvt_f32_bf16(const float* __restrict__ src, bf16* __restrict__ dst, int n4) {
    int t = blockIdx.x * 256 + threadIdx.x;
    if (t >= n4) return;
    const float4 f = reinterpret_cast<const float4*>(src)[t];
    union { bf16 h[4]; uint2 u; } o;
    o.h[0] = (bf16)f.x; o.h[1] = (bf16)f.y; o.h[2] = (bf16)f.z; o.h[3] = (bf16)f.w;
    reinterpret_cast<uint2*>(dst)[t] = o.u;
}

// --------------------------- K1: embed gather ------------------------------
__global__ void gather_embed(const int* __restrict__ ids, const float* __restrict__ emb,
                             bf16* __restrict__ xbf) {
    size_t t  = (size_t)blockIdx.x * 256 + threadIdx.x;   // B*S*(E/4) threads
    int    e4 = (int)(t & (E / 4 - 1));
    size_t bs = t >> 7;                                   // E/4 == 128
    int row = ids[bs];
    const float4 f = *reinterpret_cast<const float4*>(emb + (size_t)row * E + e4 * 4);
    union { bf16 h[4]; uint2 u; } o;
    o.h[0] = (bf16)f.x; o.h[1] = (bf16)f.y; o.h[2] = (bf16)f.z; o.h[3] = (bf16)f.w;
    *reinterpret_cast<uint2*>(xbf + bs * E + e4 * 4) = o.u;
}

// --------------------------- K2: zero state --------------------------------
__global__ void zero_bytes16(uint4* __restrict__ p) {
    size_t t = (size_t)blockIdx.x * 256 + threadIdx.x;
    p[t] = uint4{0u, 0u, 0u, 0u};
}

// ------------------ K3: xg = x @ W_ih^T  (fully parallel) ------------------
// Output xg[S*B, 4H] bf16, row r = s*B + b. Grid: 32768 WGs x 256 threads.
__global__ __launch_bounds__(256) void xg_gemm(
    const bf16* __restrict__ xbf, const bf16* __restrict__ wih,
    bf16* __restrict__ xg)
{
    const int lane = threadIdx.x & 31;
    const int wave = threadIdx.x >> 5;
    const int sub  = lane >> 4;
    const int lm   = lane & 15;
    const int wgM  = blockIdx.x >> 6;          // 0..511  (128 rows each)
    const int wgN  = blockIdx.x & 63;          // 0..63   (64 cols each)
    const int r0   = (wgM * 8 + wave) << 4;    // M-tile base row (s*B+b space)
    const int n0   = wgN << 6;                 // N base col

    const int r = r0 + lm;
    const int s = r >> 7;
    const int b = r & (B - 1);
    const bf16* arow  = xbf + ((size_t)b * S + s) * E;
    const bf16* brow0 = wih + (size_t)(n0 + 0 * 16 + lm) * E;
    const bf16* brow1 = wih + (size_t)(n0 + 1 * 16 + lm) * E;
    const bf16* brow2 = wih + (size_t)(n0 + 2 * 16 + lm) * E;
    const bf16* brow3 = wih + (size_t)(n0 + 3 * 16 + lm) * E;

    v8f a0 = {0,0,0,0,0,0,0,0}, a1 = {0,0,0,0,0,0,0,0};
    v8f a2 = {0,0,0,0,0,0,0,0}, a3 = {0,0,0,0,0,0,0,0};

    gemm_k_pipe(arow, brow0, brow1, brow2, brow3, E, sub, a0, a1, a2, a3);

    #pragma unroll
    for (int rr = 0; rr < 8; ++rr) {
        const size_t orow = (size_t)(r0 + (sub << 3) + rr) * G4H;
        xg[orow + n0 + 0 * 16 + lm] = (bf16)a0[rr];
        xg[orow + n0 + 1 * 16 + lm] = (bf16)a1[rr];
        xg[orow + n0 + 2 * 16 + lm] = (bf16)a2[rr];
        xg[orow + n0 + 3 * 16 + lm] = (bf16)a3[rr];
    }
}

// ---------------- K4a: step with precomputed xg, 2-way K-split -------------
// grid = 128 WGs: bx&63 -> 16 hidden cols, bx>>6 -> 64-row group.
// block = 256 = 8 waves: wave&3 -> row tile (16 rows), wave>>2 -> K half.
// Upper-K waves reduce their partials into LDS; lower-K waves add + epilogue.
__global__ __launch_bounds__(256) void lstm_step_xg(
    const bf16* __restrict__ xg,   const bf16* __restrict__ whh,
    const float* __restrict__ bias,
    const bf16* __restrict__ h_in, bf16* __restrict__ h_out,
    float* __restrict__ c, int t)
{
    __shared__ float4 red[4][32][9];           // 18 KB, padded stride

    const int lane   = threadIdx.x & 31;
    const int wave   = threadIdx.x >> 5;
    const int sub    = lane >> 4;
    const int lm     = lane & 15;
    const int rowTl  = wave & 3;
    const int kHalf  = wave >> 2;
    const int kOff   = kHalf * (H / 2);        // 0 or 512
    const int n0     = (blockIdx.x & 63) << 4;
    const int m0     = ((blockIdx.x >> 6) << 6) + (rowTl << 4);

    const bf16* arow_h = h_in + (size_t)(m0 + lm) * H + kOff;
    const bf16* bhh0 = whh + (size_t)(0 * H + n0 + lm) * H + kOff;
    const bf16* bhh1 = whh + (size_t)(1 * H + n0 + lm) * H + kOff;
    const bf16* bhh2 = whh + (size_t)(2 * H + n0 + lm) * H + kOff;
    const bf16* bhh3 = whh + (size_t)(3 * H + n0 + lm) * H + kOff;

    v8f acc0 = {0,0,0,0,0,0,0,0}, acc1 = {0,0,0,0,0,0,0,0};
    v8f acc2 = {0,0,0,0,0,0,0,0}, acc3 = {0,0,0,0,0,0,0,0};

    gemm_k_pipe(arow_h, bhh0, bhh1, bhh2, bhh3, H / 2, sub, acc0, acc1, acc2, acc3);

    union { v8f v; float4 q[2]; } u;
    if (kHalf) {                               // upper half: spill partials
        float4* dst = red[rowTl][lane];
        u.v = acc0; dst[0] = u.q[0]; dst[1] = u.q[1];
        u.v = acc1; dst[2] = u.q[0]; dst[3] = u.q[1];
        u.v = acc2; dst[4] = u.q[0]; dst[5] = u.q[1];
        u.v = acc3; dst[6] = u.q[0]; dst[7] = u.q[1];
    }
    __syncthreads();
    if (kHalf) return;

    const float4* src = red[rowTl][lane];      // lower half: combine + cell
    u.q[0] = src[0]; u.q[1] = src[1]; acc0 += u.v;
    u.q[0] = src[2]; u.q[1] = src[3]; acc1 += u.v;
    u.q[0] = src[4]; u.q[1] = src[5]; acc2 += u.v;
    u.q[0] = src[6]; u.q[1] = src[7]; acc3 += u.v;

    const int col = n0 + lm;
    const float bi = bias[0 * H + col];
    const float bf = bias[1 * H + col];
    const float bg = bias[2 * H + col];
    const float bo = bias[3 * H + col];
    const bf16* xgt = xg + (size_t)t * B * G4H;

    #pragma unroll
    for (int r = 0; r < 8; ++r) {
        const int brow = m0 + (sub << 3) + r;
        const size_t idx  = (size_t)brow * H + col;
        const size_t xrow = (size_t)brow * G4H;
        const float gi = acc0[r] + (float)xgt[xrow + 0 * H + col] + bi;
        const float gf = acc1[r] + (float)xgt[xrow + 1 * H + col] + bf;
        const float gg = acc2[r] + (float)xgt[xrow + 2 * H + col] + bg;
        const float go = acc3[r] + (float)xgt[xrow + 3 * H + col] + bo;
        const float cn = sigf(gf) * c[idx] + sigf(gi) * tanhf(gg);
        const float hn = sigf(go) * tanhf(cn);
        c[idx] = cn;
        h_out[idx] = (bf16)hn;
    }
}

// ---------------- K4b: fused step, fallback (K = 1536) ---------------------
__global__ __launch_bounds__(256) void lstm_step_fused(
    const bf16* __restrict__ xbf,  const bf16* __restrict__ wih,
    const bf16* __restrict__ whh,  const float* __restrict__ bias,
    const bf16* __restrict__ h_in, bf16* __restrict__ h_out,
    float* __restrict__ c, int t)
{
    const int lane = threadIdx.x & 31;
    const int wave = threadIdx.x >> 5;
    const int sub  = lane >> 4;
    const int lm   = lane & 15;
    const int m0   = wave << 4;
    const int n0   = blockIdx.x << 4;

    const bf16* arow_x = xbf + ((size_t)(m0 + lm) * S + t) * E;
    const bf16* arow_h = h_in + (size_t)(m0 + lm) * H;
    const bf16* bih0 = wih + (size_t)(0 * H + n0 + lm) * E;
    const bf16* bih1 = wih + (size_t)(1 * H + n0 + lm) * E;
    const bf16* bih2 = wih + (size_t)(2 * H + n0 + lm) * E;
    const bf16* bih3 = wih + (size_t)(3 * H + n0 + lm) * E;
    const bf16* bhh0 = whh + (size_t)(0 * H + n0 + lm) * H;
    const bf16* bhh1 = whh + (size_t)(1 * H + n0 + lm) * H;
    const bf16* bhh2 = whh + (size_t)(2 * H + n0 + lm) * H;
    const bf16* bhh3 = whh + (size_t)(3 * H + n0 + lm) * H;

    v8f acc0 = {0,0,0,0,0,0,0,0}, acc1 = {0,0,0,0,0,0,0,0};
    v8f acc2 = {0,0,0,0,0,0,0,0}, acc3 = {0,0,0,0,0,0,0,0};

    gemm_k_pipe(arow_x, bih0, bih1, bih2, bih3, E, sub, acc0, acc1, acc2, acc3);
    gemm_k_pipe(arow_h, bhh0, bhh1, bhh2, bhh3, H, sub, acc0, acc1, acc2, acc3);

    const int col = n0 + lm;
    const float bi = bias[0 * H + col];
    const float bf = bias[1 * H + col];
    const float bg = bias[2 * H + col];
    const float bo = bias[3 * H + col];

    #pragma unroll
    for (int r = 0; r < 8; ++r) {
        const int brow = m0 + (sub << 3) + r;
        const size_t idx = (size_t)brow * H + col;
        const float gi = acc0[r] + bi;
        const float gf = acc1[r] + bf;
        const float gg = acc2[r] + bg;
        const float go = acc3[r] + bo;
        const float cn = sigf(gf) * c[idx] + sigf(gi) * tanhf(gg);
        const float hn = sigf(go) * tanhf(cn);
        c[idx] = cn;
        h_out[idx] = (bf16)hn;
    }
}

// --------------------------- K5: classifier --------------------------------
__global__ void classifier(const bf16* __restrict__ h, const float* __restrict__ wout,
                           const float* __restrict__ bout, float* __restrict__ out) {
    int t = blockIdx.x * 256 + threadIdx.x;   // B*OUTN = 1024 threads
    int b = t >> 3;
    int o = t & (OUTN - 1);
    const bf16*  hr = h + (size_t)b * H;
    const float* wr = wout + (size_t)o * H;
    float s = bout[o];
    #pragma unroll 8
    for (int k = 0; k < H; ++k) s += (float)hr[k] * wr[k];
    out[(size_t)b * OUTN + o] = s;
}

// ---------------------------------------------------------------------------
extern "C" void kernel_launch(void* const* d_in, const int* in_sizes, int n_in,
                              void* d_out, int out_size, void* d_ws, size_t ws_size,
                              hipStream_t stream) {
    const int*   ids   = (const int*)  d_in[0];
    const float* emb   = (const float*)d_in[1];
    const float* W_ih  = (const float*)d_in[2];
    const float* W_hh  = (const float*)d_in[3];
    const float* bias  = (const float*)d_in[4];
    const float* W_out = (const float*)d_in[5];
    const float* b_out = (const float*)d_in[6];
    float* out = (float*)d_out;

    // Workspace layout:
    //   [wih_bf 4M][whh_bf 8M][x_bf 64M][h_bf 2x256K][c 512K]  = 77 MiB
    //   [xg 512M]  (optional, Path A)                          = 589 MiB total
    char* ws = (char*)d_ws;
    const size_t MiB = 1024 * 1024;
    bf16*  wih_bf = (bf16*)(ws);
    bf16*  whh_bf = (bf16*)(ws + 4 * MiB);
    bf16*  xbf    = (bf16*)(ws + 12 * MiB);
    bf16*  hbf    = (bf16*)(ws + 76 * MiB);
    float* cbuf   = (float*)(ws + 76 * MiB + 512 * 1024);
    bf16*  xg     = (bf16*)(ws + 77 * MiB);
    const bool use_xg = ws_size >= 589 * MiB;

    // K0: weight conversion
    cvt_f32_bf16<<<(G4H * E / 4 + 255) / 256, 256, 0, stream>>>(W_ih, wih_bf, G4H * E / 4);
    cvt_f32_bf16<<<(G4H * H / 4 + 255) / 256, 256, 0, stream>>>(W_hh, whh_bf, G4H * H / 4);
    // K1: embedding gather
    gather_embed<<<(int)(((size_t)B * S * (E / 4)) / 256), 256, 0, stream>>>(ids, emb, xbf);
    // K2: zero h (both buffers) + c : contiguous 1 MiB
    zero_bytes16<<<256, 256, 0, stream>>>((uint4*)hbf);

    const size_t HB = (size_t)B * H;
    if (use_xg) {
        // K3: hoist input projection off the sequential critical path
        xg_gemm<<<32768, 256, 0, stream>>>(xbf, wih_bf, xg);
        // K4a: 512 dependent steps, K=1024 split across wave pairs
        for (int t = 0; t < S; ++t) {
            const bf16* h_in  = hbf + (size_t)(t & 1) * HB;
            bf16*       h_out = hbf + (size_t)((t + 1) & 1) * HB;
            lstm_step_xg<<<128, 256, 0, stream>>>(xg, whh_bf, bias, h_in, h_out, cbuf, t);
        }
    } else {
        // K4b: fused steps, K=1536 each
        for (int t = 0; t < S; ++t) {
            const bf16* h_in  = hbf + (size_t)(t & 1) * HB;
            bf16*       h_out = hbf + (size_t)((t + 1) & 1) * HB;
            lstm_step_fused<<<H / 16, 256, 0, stream>>>(xbf, wih_bf, whh_bf, bias,
                                                        h_in, h_out, cbuf, t);
        }
    }
    // K5: classifier on h_last (buffer (S&1) == 0)
    classifier<<<(B * OUTN) / 256, 256, 0, stream>>>(hbf, W_out, b_out, out);
}
